// MultiAttention_1932735283933
// MI455X (gfx1250) — matrix operand order repaired and checked by reading, not compiled, not run on hardware
//
#include <hip/hip_runtime.h>
#include <hip/hip_bf16.h>
#include <math.h>

// ---------------------------------------------------------------------------
// Shapes (fixed by the reference)
// ---------------------------------------------------------------------------
#define BDIM   1024          // BERT_DIM
#define HDIM   512           // H_DIM
#define NB     64            // batch
#define LL     512           // seq len
#define MROWS  (NB * LL)     // 32768 GEMM rows
#define MBLK   128           // rows per block (8 waves * 16)

// Packed-fragment LDS layout: per (tile, kt) a 1 KB region:
//   [0,512)    lo 16B chunks, lane-sequential (elements 0..7 of v16h)
//   [512,1024) hi 16B chunks, lane-sequential (elements 8..15 of v16h)
// -> every ds_load_b128 across a wave is a sequential 512B run: conflict-free,
//    and all fragment addresses are (per-lane base + 16-bit immediate offset).
#define A_LDS_BYTES  (8 * 32 * 1024)       // 8 m-tiles x 32 kt x 1KB = 256 KB
#define B_LDS_BYTES  (4 * 8 * 1024)        // 4 n-tiles x 8 kt  x 1KB =  32 KB

typedef _Float16 v8h  __attribute__((ext_vector_type(8)));
typedef _Float16 v16h __attribute__((ext_vector_type(16)));
typedef _Float16 h4   __attribute__((ext_vector_type(4)));
typedef float    v8f  __attribute__((ext_vector_type(8)));

// ---------------------------------------------------------------------------
// Branchless tanh: 1 - 2/(exp(2x)+1), clamp so exp can't overflow.
// ---------------------------------------------------------------------------
__device__ __forceinline__ float tanh_fast(float x) {
    float cx = fminf(fmaxf(x, -9.0f), 9.0f);
    float e  = __expf(2.0f * cx);
    return 1.0f - 2.0f / (e + 1.0f);
}

// ---------------------------------------------------------------------------
// Kernel 0a: pack W1s (rows 0..1023 of W1, f32 [k][n]) into WMMA-B fragment
// order, f16:  Wp[(nt*32+kt)*1024 + halfSel*512 + lane*16 + j*2]
// ---------------------------------------------------------------------------
__global__ void k_pack_b(const float* __restrict__ W1, _Float16* __restrict__ Wp) {
    int c = blockIdx.x * 256 + threadIdx.x;        // 16B chunk index, 0..65535
    int lane = c & 31;
    int half = (c >> 5) & 1;
    int kt   = (c >> 6) & 31;
    int nt   = c >> 11;
    int n  = nt * 16 + (lane & 15);
    int k0 = kt * 32 + ((lane >> 4) * 8) + half * 16;
    v8h v;
#pragma unroll
    for (int j = 0; j < 8; ++j)
        v[j] = (_Float16)W1[(size_t)(k0 + j) * HDIM + n];
    *(v8h*)(Wp + (size_t)c * 8) = v;
}

// ---------------------------------------------------------------------------
// Kernel 0b: pproj[b][n] = p[b,:] @ W1p[:,n] + b1[n]   (64x512, tiny)
// ---------------------------------------------------------------------------
__global__ void k_pproj(const float* __restrict__ p, const float* __restrict__ W1,
                        const float* __restrict__ b1, float* __restrict__ pproj) {
    int gid = blockIdx.x * 256 + threadIdx.x;      // 0..32767
    int b = gid >> 9;
    int n = gid & 511;
    const float* pr = p + b * BDIM;
    const float* w  = W1 + (size_t)BDIM * HDIM + n;
    float acc = b1[n];
#pragma unroll 4
    for (int k = 0; k < BDIM; ++k)
        acc = fmaf(pr[k], w[(size_t)k * HDIM], acc);
    pproj[gid] = acc;
}

// ---------------------------------------------------------------------------
// Kernel 1: fused GEMM + tanh + dot(w2) -> scores.
// 8 waves x 16 rows = 128 rows/block. 4 N-tile register blocking per wave.
// B slices stream through LDS via the CDNA5 async global->LDS engine.
// ---------------------------------------------------------------------------
__global__ void k_scores(const float* __restrict__ sent, const _Float16* __restrict__ Wp,
                         const float* __restrict__ pproj, const float* __restrict__ w2,
                         const float* __restrict__ b2, float* __restrict__ scores) {
    extern __shared__ __align__(16) char smem_raw[];
    char* As = smem_raw;                           // 256 KB packed A fragments
    char* Bs = smem_raw + A_LDS_BYTES;             //  32 KB packed B fragments

    const int tid     = threadIdx.x;               // 0..255
    const int lane    = tid & 31;
    const int wave    = tid >> 5;                  // 0..7  (= m-tile)
    const int nl      = lane & 15;
    const int rowBase = blockIdx.x * MBLK;
    const int bIdx    = rowBase >> 9;              // batch (512 rows/batch)

    // ---- stage A panel: 128 rows x 1024 f32 -> f16 packed fragments ----
    {
        const float4* sv = (const float4*)(sent + (size_t)rowBase * BDIM);
#pragma unroll 4
        for (int i = 0; i < 128; ++i) {
            int linear = i * 256 + tid;            // float4 index (m, k/4)
            int m  = linear >> 8;
            int k  = (linear & 255) * 4;
            int kt = k >> 5;
            int kk = k & 31;
            int laneHi = (kk >> 3) & 1;            // which lane half owns this run
            int hiChnk = (kk >> 4) & 1;            // lo/hi 512B chunk
            float4 v = sv[linear];
            h4 hv = { (_Float16)v.x, (_Float16)v.y, (_Float16)v.z, (_Float16)v.w };
            char* dst = As + (((m >> 4) * 32 + kt) * 1024)
                           + hiChnk * 512
                           + (laneHi * 16 + (m & 15)) * 16
                           + (kk & 7) * 2;
            *(h4*)dst = hv;
        }
    }
    __syncthreads();

    // Per-lane bases: every fragment below is base + (constant <= 32KB), which
    // folds into the 16-bit DS immediate offset field (no per-fragment v_add).
    const char* aLane = As + wave * (32 * 1024) + lane * 16;
    const char* bLane = Bs + lane * 16;

    float sAcc[8] = {0.f, 0.f, 0.f, 0.f, 0.f, 0.f, 0.f, 0.f};

    for (int g = 0; g < 8; ++g) {                  // N-tile groups of 4 (32 tiles)
        v8f acc[4] = {};
        for (int kq = 0; kq < 4; ++kq) {           // K quarters (8 kt each)
            __syncthreads();                       // B_s readers from prev stage done
            // ---- async-stage B: 4 contiguous 8KB segments from Wp ----
            {
#pragma unroll
                for (int i = 0; i < 8; ++i) {
                    int c   = i * 256 + tid;       // 16B chunk, 0..2047
                    int ntL = c >> 9;
                    int w16 = c & 511;
                    const char* src = (const char*)Wp
                        + ((size_t)((g * 4 + ntL) * 32 + kq * 8) * 1024) + w16 * 16;
                    unsigned dstOff = (unsigned)(size_t)(Bs + ntL * 8192 + w16 * 16);
                    asm volatile("global_load_async_to_lds_b128 %0, %1, off"
                                 :: "v"(dstOff), "v"(src) : "memory");
                }
                asm volatile("s_wait_asynccnt 0x0" ::: "memory");
            }
            __syncthreads();

            const char* aK = aLane + kq * (8 * 1024);
#pragma unroll
            for (int kt = 0; kt < 8; ++kt) {
                v8h alo = *(const v8h*)(aK + kt * 1024);
                v8h ahi = *(const v8h*)(aK + kt * 1024 + 512);
                v16h a  = __builtin_shufflevector(alo, ahi,
                           0,1,2,3,4,5,6,7,8,9,10,11,12,13,14,15);
#pragma unroll
                for (int nL = 0; nL < 4; ++nL) {
                    v8h blo = *(const v8h*)(bLane + (nL * 8 + kt) * 1024);
                    v8h bhi = *(const v8h*)(bLane + (nL * 8 + kt) * 1024 + 512);
                    v16h bb = __builtin_shufflevector(blo, bhi,
                               0,1,2,3,4,5,6,7,8,9,10,11,12,13,14,15);
                    acc[nL] = __builtin_amdgcn_wmma_f32_16x16x32_f16(
                        false, a, false, bb, (short)0, acc[nL], false, false);
                }
            }
        }

        // ---- fused epilogue: tanh + dot with w2, reduce over 16 N lanes ----
#pragma unroll
        for (int nL = 0; nL < 4; ++nL) {
            const int n     = (g * 4 + nL) * 16 + nl;
            const float w2n = w2[n];
            const float pbn = pproj[bIdx * HDIM + n];
#pragma unroll
            for (int r = 0; r < 8; ++r) {
                // C layout: lanes 0-15 -> M=r, lanes 16-31 -> M=8+r, N = lane&15
                float v = tanh_fast(acc[nL][r] + pbn) * w2n;
                v += __shfl_xor(v, 1, 32);
                v += __shfl_xor(v, 2, 32);
                v += __shfl_xor(v, 4, 32);
                v += __shfl_xor(v, 8, 32);
                sAcc[r] += v;
            }
        }
    }

    if (nl == 0) {                                 // lanes 0 and 16 write 8 rows each
        const float b2v = b2[0];
        int mBase = rowBase + wave * 16 + (lane >> 4) * 8;
#pragma unroll
        for (int r = 0; r < 8; ++r)
            scores[mBase + r] = sAcc[r] + b2v;
    }
}

// ---------------------------------------------------------------------------
// Kernel 2: masked softmax over L=512, both masks. One block per batch.
// ---------------------------------------------------------------------------
__device__ __forceinline__ void softmax512(float s0, float s1, bool m0, bool m1,
                                           float* __restrict__ out, int tid,
                                           float* __restrict__ red) {
    float v0 = m0 ? s0 : -INFINITY;
    float v1 = m1 ? s1 : -INFINITY;
    float mx = fmaxf(v0, v1);
#pragma unroll
    for (int o = 1; o < 32; o <<= 1) mx = fmaxf(mx, __shfl_xor(mx, o, 32));
    if ((tid & 31) == 0) red[tid >> 5] = mx;
    __syncthreads();
    float bmx = red[0];
#pragma unroll
    for (int i = 1; i < 8; ++i) bmx = fmaxf(bmx, red[i]);
    __syncthreads();
    float e0 = expf(v0 - bmx);
    float e1 = expf(v1 - bmx);
    float sum = e0 + e1;
#pragma unroll
    for (int o = 1; o < 32; o <<= 1) sum += __shfl_xor(sum, o, 32);
    if ((tid & 31) == 0) red[tid >> 5] = sum;
    __syncthreads();
    float ts = 0.f;
#pragma unroll
    for (int i = 0; i < 8; ++i) ts += red[i];
    __syncthreads();
    float inv = 1.0f / ts;
    out[tid]       = e0 * inv;
    out[tid + 256] = e1 * inv;
}

__global__ void k_softmax(const float* __restrict__ scores,
                          const unsigned char* __restrict__ am,
                          const unsigned char* __restrict__ bm,
                          float* __restrict__ aw, float* __restrict__ bw) {
    __shared__ float red[8];
    const int b = blockIdx.x, tid = threadIdx.x;
    const float* s = scores + b * LL;
    float s0 = s[tid], s1 = s[tid + 256];
    const unsigned char* amb = am + b * LL;
    const unsigned char* bmb = bm + b * LL;
    softmax512(s0, s1, amb[tid] != 0, amb[tid + 256] != 0, aw + b * LL, tid, red);
    softmax512(s0, s1, bmb[tid] != 0, bmb[tid + 256] != 0, bw + b * LL, tid, red);
}

// ---------------------------------------------------------------------------
// Kernel 3: a_out[b,d] = sum_l aw[b,l]*sent[b,l,d]  (and b_out). Streaming pass.
// ---------------------------------------------------------------------------
__global__ void k_pool(const float* __restrict__ sent, const float* __restrict__ aw,
                       const float* __restrict__ bw, float* __restrict__ out) {
    __shared__ float wA[LL];
    __shared__ float wB[LL];
    const int b   = blockIdx.y;
    const int tid = threadIdx.x;
    const int d   = blockIdx.x * 256 + tid;
    for (int i = tid; i < LL; i += 256) {
        wA[i] = aw[b * LL + i];
        wB[i] = bw[b * LL + i];
    }
    __syncthreads();
    const float* sb = sent + (size_t)b * LL * BDIM + d;
    float accA = 0.f, accB = 0.f;
#pragma unroll 4
    for (int l = 0; l < LL; ++l) {
        float s = sb[(size_t)l * BDIM];
        accA = fmaf(wA[l], s, accA);
        accB = fmaf(wB[l], s, accB);
    }
    out[b * BDIM + d]             = accA;          // a_out
    out[NB * BDIM + b * BDIM + d] = accB;          // b_out (concatenated)
}

// ---------------------------------------------------------------------------
// Host-side launch (graph-capture safe: stream-only, no alloc/sync)
// ---------------------------------------------------------------------------
extern "C" void kernel_launch(void* const* d_in, const int* in_sizes, int n_in,
                              void* d_out, int out_size, void* d_ws, size_t ws_size,
                              hipStream_t stream) {
    const float*         sent = (const float*)d_in[0];          // (64,512,1024) f32
    const float*         p    = (const float*)d_in[1];          // (64,1024) f32
    const unsigned char* am   = (const unsigned char*)d_in[2];  // (64,512) bool
    const unsigned char* bm   = (const unsigned char*)d_in[3];  // (64,512) bool
    const float*         W1   = (const float*)d_in[4];          // (2048,512) f32
    const float*         b1   = (const float*)d_in[5];          // (512,) f32
    const float*         w2   = (const float*)d_in[6];          // (512,) f32
    const float*         b2   = (const float*)d_in[7];          // scalar f32
    float* out = (float*)d_out;                                  // 2*(64,1024) f32

    // workspace layout
    char* ws = (char*)d_ws;
    _Float16* Wp  = (_Float16*)ws;                              // 1 MB packed W1s f16
    float* pproj  = (float*)(ws + (1u << 20));                  // 128 KB
    float* scores = (float*)(ws + (1u << 20) + (128u << 10));   // 128 KB
    float* aw     = (float*)(ws + (1u << 20) + (256u << 10));   // 128 KB
    float* bw     = (float*)(ws + (1u << 20) + (384u << 10));   // 128 KB
    (void)in_sizes; (void)n_in; (void)out_size; (void)ws_size;

    k_pack_b<<<(32 * 32 * 64) / 256, 256, 0, stream>>>(W1, Wp);
    k_pproj<<<(NB * HDIM) / 256, 256, 0, stream>>>(p, W1, b1, pproj);

    const size_t smemBytes = A_LDS_BYTES + B_LDS_BYTES;          // 288 KB
    k_scores<<<MROWS / MBLK, 256, smemBytes, stream>>>(sent, Wp, pproj, w2, b2, scores);

    k_softmax<<<NB, 256, 0, stream>>>(scores, am, bm, aw, bw);
    k_pool<<<dim3(BDIM / 256, NB), 256, 0, stream>>>(sent, aw, bw, out);
}
